// ChessGATv2_63857573757019
// MI455X (gfx1250) — matrix operand order repaired and checked by reading, not compiled, not run on hardware
//
#include <hip/hip_runtime.h>
#include <hip/hip_bf16.h>

#define N_NODES 16384
#define N_EDGES 393216
#define EF      (N_EDGES + N_NODES)   // 409600 (edges + self loops)
#define NGRAPH  256
#define HID     128
#define NHEAD   4
#define HC      32
#define NLAYER  8
#define NODE_DIM 21
#define EDGE_DIM 6

typedef __bf16 bf16;
typedef bf16  v16bf __attribute__((ext_vector_type(16)));
typedef bf16  v8bf  __attribute__((ext_vector_type(8)));
typedef float v8f   __attribute__((ext_vector_type(8)));

union AFrag { v16bf v; v8bf h[2]; };

__device__ __forceinline__ void atomicMaxF(float* addr, float val) {
    // standard sign-split trick: positives order as ints, negatives reverse as uints
    if (val >= 0.0f) atomicMax((int*)addr, __float_as_int(val));
    else             atomicMin((unsigned int*)addr, (unsigned int)__float_as_int(val));
}

// ---------------------------------------------------------------- prep kernels
__global__ void k_convert_weights(const float* __restrict__ Wl, const float* __restrict__ Wr,
                                  bf16* __restrict__ Wlb, bf16* __restrict__ Wrb, int n) {
    int i = blockIdx.x * blockDim.x + threadIdx.x;
    if (i >= n) return;
    Wlb[i] = (bf16)Wl[i];
    Wrb[i] = (bf16)Wr[i];
}

__global__ void k_prep_init(float* __restrict__ deg, float* __restrict__ lattr,
                            float* __restrict__ pooled, float* __restrict__ cnt) {
    int i = blockIdx.x * blockDim.x + threadIdx.x;
    if (i < N_NODES * EDGE_DIM) lattr[i] = 0.0f;
    if (i < N_NODES)            deg[i]   = 0.0f;
    if (i < NGRAPH * HID)       pooled[i] = 0.0f;
    if (i < NGRAPH)             cnt[i]   = 0.0f;
}

__global__ void k_deg_loop(const int* __restrict__ dst0, const float* __restrict__ eattr,
                           float* __restrict__ deg, float* __restrict__ lattr) {
    int e = blockIdx.x * blockDim.x + threadIdx.x;
    if (e >= N_EDGES) return;
    int d = dst0[e];
    atomicAdd(&deg[d], 1.0f);
    #pragma unroll
    for (int k = 0; k < EDGE_DIM; ++k)
        atomicAdd(&lattr[(size_t)d * EDGE_DIM + k], eattr[(size_t)e * EDGE_DIM + k]);
}

__global__ void k_loop_div(float* __restrict__ lattr, const float* __restrict__ deg) {
    int i = blockIdx.x * blockDim.x + threadIdx.x;
    if (i >= N_NODES * EDGE_DIM) return;
    lattr[i] /= fmaxf(deg[i / EDGE_DIM], 1.0f);
}

__global__ void k_input_linear(const float* __restrict__ x, const float* __restrict__ Win,
                               const float* __restrict__ b_in, float* __restrict__ h) {
    int i = blockIdx.x * blockDim.x + threadIdx.x;
    if (i >= N_NODES * HID) return;
    int n = i >> 7, j = i & (HID - 1);
    const float* xr = x + (size_t)n * NODE_DIM;
    float s = b_in[j];
    #pragma unroll
    for (int k = 0; k < NODE_DIM; ++k) s += xr[k] * Win[k * HID + j];
    h[i] = s > 0.0f ? s : 0.0f;
}

// ---------------------------------------------------------------- per-layer
__global__ void k_layer_init(float* __restrict__ outb, float* __restrict__ smax,
                             float* __restrict__ den, float* __restrict__ stats) {
    int i = blockIdx.x * blockDim.x + threadIdx.x;
    if (i < N_NODES * HID)   outb[i] = 0.0f;
    if (i < N_NODES * NHEAD) { den[i] = 0.0f; smax[i] = -__builtin_huge_valf(); }
    if (i < 2 * HID)         stats[i] = 0.0f;
}

__global__ void k_h2bf16(const float* __restrict__ h, bf16* __restrict__ hb,
                         float* __restrict__ res, int saveRes) {
    int i = blockIdx.x * blockDim.x + threadIdx.x;
    if (i >= N_NODES * HID) return;
    float v = h[i];
    hb[i] = (bf16)v;
    if (saveRes) res[i] = v;
}

// xl = h@Wl + bl ; xr = h@Wr + br  via bf16 WMMA, fp32 accumulate.
// Block = 256 threads (8 wave32), handles 16 rows; 16 jobs = {xl,xr} x 8 col-tiles.
__global__ void k_node_linear(const bf16* __restrict__ hb,
                              const bf16* __restrict__ Wlb, const bf16* __restrict__ Wrb,
                              const float* __restrict__ bl, const float* __restrict__ br,
                              float* __restrict__ xl, float* __restrict__ xr) {
    const int lane = threadIdx.x & 31;
    const int wave = threadIdx.x >> 5;
    const int row0 = blockIdx.x << 4;
    const int hi   = lane >> 4;        // which half of the wave
    const int lo   = lane & 15;
    const int arow = row0 + lo;        // A-operand row held by this lane

    for (int job = wave; job < 16; job += 8) {
        const int   mat  = job >> 3;
        const int   ct   = job & 7;                    // 16-col output tile
        const bf16* Wb   = mat ? Wrb : Wlb;
        const float* bias = mat ? br : bl;
        float*      xo   = mat ? xr : xl;

        v8f acc = {};
        #pragma unroll
        for (int ks = 0; ks < 4; ++ks) {               // K = 128 in 4 x 32 steps
            // A 16x32 bf16: lane m holds K = {hi*8+0..7, 16+hi*8+0..7}
            AFrag a;
            const bf16* ap = hb + (size_t)arow * HID + ks * 32 + hi * 8;
            a.h[0] = *(const v8bf*)(ap);
            a.h[1] = *(const v8bf*)(ap + 16);
            // B 32x16 bf16: lane n = lo holds K = {hi*8+0..7, 16+hi*8+0..7} (strided gather)
            v16bf b;
            #pragma unroll
            for (int s = 0; s < 16; ++s)
                b[s] = Wb[(size_t)(ks * 32 + (s < 8 ? hi * 8 + s : 16 + hi * 8 + (s - 8))) * HID + ct * 16 + lo];
            acc = __builtin_amdgcn_wmma_f32_16x16x32_bf16(
                false, a.v, false, b, (short)0, acc, false, false);
        }
        const int col = ct * 16 + lo;
        const float bv = bias[col];
        #pragma unroll
        for (int r = 0; r < 8; ++r)                    // D: VGPR r -> row hi*8+r
            xo[(size_t)(row0 + hi * 8 + r) * HID + col] = acc[r] + bv;
    }
}

// per-edge attention logits s[e][h] + segment max (per dst)
__global__ void k_edge_score(const int* __restrict__ src0, const int* __restrict__ dst0,
                             const float* __restrict__ eattr, const float* __restrict__ lattr,
                             const float* __restrict__ xl, const float* __restrict__ xr,
                             const float* __restrict__ We, const float* __restrict__ att,
                             float* __restrict__ sbuf, float* __restrict__ smax) {
    __shared__ float sWe[EDGE_DIM * HID];
    __shared__ float sAtt[HID];
    for (int i = threadIdx.x; i < EDGE_DIM * HID; i += blockDim.x) sWe[i] = We[i];
    for (int i = threadIdx.x; i < HID; i += blockDim.x)            sAtt[i] = att[i];
    __syncthreads();

    int e = blockIdx.x * blockDim.x + threadIdx.x;
    if (e >= EF) return;
    int es, ed; const float* ea;
    if (e < N_EDGES) { es = src0[e]; ed = dst0[e]; ea = eattr + (size_t)e * EDGE_DIM; }
    else             { es = ed = e - N_EDGES;      ea = lattr + (size_t)es * EDGE_DIM; }
    const float a0 = ea[0], a1 = ea[1], a2 = ea[2], a3 = ea[3], a4 = ea[4], a5 = ea[5];
    const float* xlp = xl + (size_t)es * HID;
    const float* xrp = xr + (size_t)ed * HID;

    #pragma unroll
    for (int hh = 0; hh < NHEAD; ++hh) {
        float acc = 0.0f;
        #pragma unroll 8
        for (int c = 0; c < HC; ++c) {
            int j = hh * HC + c;
            float ee = a0 * sWe[j] + a1 * sWe[HID + j] + a2 * sWe[2 * HID + j]
                     + a3 * sWe[3 * HID + j] + a4 * sWe[4 * HID + j] + a5 * sWe[5 * HID + j];
            float m = xlp[j] + xrp[j] + ee;
            m = m > 0.0f ? m : 0.2f * m;               // leaky_relu(0.2)
            acc += m * sAtt[j];
        }
        sbuf[(size_t)e * NHEAD + hh] = acc;
        atomicMaxF(&smax[ed * NHEAD + hh], acc);
    }
}

__global__ void k_edge_exp(const int* __restrict__ dst0, float* __restrict__ sbuf,
                           const float* __restrict__ smax, float* __restrict__ den) {
    int i = blockIdx.x * blockDim.x + threadIdx.x;
    if (i >= EF * NHEAD) return;
    int e = i >> 2, hh = i & 3;
    int ed = e < N_EDGES ? dst0[e] : e - N_EDGES;
    float ex = __expf(sbuf[i] - smax[ed * NHEAD + hh]);
    sbuf[i] = ex;
    atomicAdd(&den[ed * NHEAD + hh], ex);
}

__global__ void k_edge_aggr(const int* __restrict__ src0, const int* __restrict__ dst0,
                            const float* __restrict__ sbuf, const float* __restrict__ den,
                            const float* __restrict__ xl, float* __restrict__ outb) {
    int i = blockIdx.x * blockDim.x + threadIdx.x;
    if (i >= EF * NHEAD) return;
    int e = i >> 2, hh = i & 3;
    int es, ed;
    if (e < N_EDGES) { es = src0[e]; ed = dst0[e]; }
    else             { es = ed = e - N_EDGES; }
    float alpha = sbuf[i] / den[ed * NHEAD + hh];
    const float* xlp = xl + (size_t)es * HID + hh * HC;
    float* op = outb + (size_t)ed * HID + hh * HC;
    #pragma unroll
    for (int c = 0; c < HC; ++c) atomicAdd(op + c, alpha * xlp[c]);
}

__global__ void k_bn_stats(const float* __restrict__ outb, float* __restrict__ stats) {
    int j = threadIdx.x;               // 128 columns
    int r0 = blockIdx.x * 128;         // 128 blocks of 128 rows
    float s1 = 0.0f, s2 = 0.0f;
    for (int r = 0; r < 128; ++r) {
        float v = outb[(size_t)(r0 + r) * HID + j];
        s1 += v; s2 += v * v;
    }
    atomicAdd(&stats[j], s1);
    atomicAdd(&stats[HID + j], s2);
}

__global__ void k_bn_finalize(float* __restrict__ stats) {
    int j = threadIdx.x;
    float mu  = stats[j] * (1.0f / N_NODES);
    float var = stats[HID + j] * (1.0f / N_NODES) - mu * mu;
    stats[2 * HID + j] = mu;
    stats[3 * HID + j] = rsqrtf(var + 1e-5f);
}

// h = relu(gamma*(out-mu)*rstd + beta [+ res])   (bconv cancels through BN)
__global__ void k_bn_apply(const float* __restrict__ outb, const float* __restrict__ stats,
                           const float* __restrict__ gamma, const float* __restrict__ beta,
                           const float* __restrict__ res, float* __restrict__ h, int addRes) {
    int i = blockIdx.x * blockDim.x + threadIdx.x;
    if (i >= N_NODES * HID) return;
    int j = i & (HID - 1);
    float y = gamma[j] * (outb[i] - stats[2 * HID + j]) * stats[3 * HID + j] + beta[j];
    if (addRes) y += res[i];
    h[i] = y > 0.0f ? y : 0.0f;
}

// ---------------------------------------------------------------- readout
__global__ void k_pool_cnt(const int* __restrict__ batch, float* __restrict__ cnt) {
    int i = blockIdx.x * blockDim.x + threadIdx.x;
    if (i >= N_NODES) return;
    atomicAdd(&cnt[batch[i]], 1.0f);
}

__global__ void k_pool_sum(const int* __restrict__ batch, const float* __restrict__ h,
                           float* __restrict__ pooled) {
    int i = blockIdx.x * blockDim.x + threadIdx.x;
    if (i >= N_NODES * HID) return;
    int n = i >> 7;
    atomicAdd(&pooled[(size_t)batch[n] * HID + (i & (HID - 1))], h[i]);
}

__global__ void k_mlp(const float* __restrict__ pooled, const float* __restrict__ cnt,
                      const float* __restrict__ W1, const float* __restrict__ b1,
                      const float* __restrict__ W2, const float* __restrict__ b2,
                      float* __restrict__ out) {
    __shared__ float pl[HID];
    __shared__ float hd[64];
    int g = blockIdx.x, t = threadIdx.x;     // 64 threads / graph
    float inv = 1.0f / fmaxf(cnt[g], 1.0f);
    for (int k = t; k < HID; k += 64) pl[k] = pooled[(size_t)g * HID + k] * inv;
    __syncthreads();
    float s = b1[t];
    #pragma unroll 8
    for (int k = 0; k < HID; ++k) s += pl[k] * W1[k * 64 + t];
    hd[t] = s > 0.0f ? s : 0.0f;
    __syncthreads();
    if (t < 3) {
        float o = b2[t];
        #pragma unroll 8
        for (int k = 0; k < 64; ++k) o += hd[k] * W2[k * 3 + t];
        out[g * 3 + t] = o;
    }
}

// ---------------------------------------------------------------- host launch
extern "C" void kernel_launch(void* const* d_in, const int* in_sizes, int n_in,
                              void* d_out, int out_size, void* d_ws, size_t ws_size,
                              hipStream_t stream) {
    const float* x     = (const float*)d_in[0];
    const int*   ei    = (const int*)d_in[1];
    const float* eattr = (const float*)d_in[2];
    const int*   batch = (const int*)d_in[3];
    const float* Win   = (const float*)d_in[4];
    const float* b_in  = (const float*)d_in[5];
    const float* Wl    = (const float*)d_in[6];
    const float* bl    = (const float*)d_in[7];
    const float* Wr    = (const float*)d_in[8];
    const float* br    = (const float*)d_in[9];
    const float* We    = (const float*)d_in[10];
    const float* att   = (const float*)d_in[11];
    // d_in[12] = bconv: provably no effect through BatchNorm (constant column shift)
    const float* gamma = (const float*)d_in[13];
    const float* beta  = (const float*)d_in[14];
    const float* W1    = (const float*)d_in[15];
    const float* b1v   = (const float*)d_in[16];
    const float* W2    = (const float*)d_in[17];
    const float* b2v   = (const float*)d_in[18];
    const int* src0 = ei;
    const int* dst0 = ei + N_EDGES;

    // workspace carve-out (floats, 64-elem aligned chunks)
    float* W = (float*)d_ws;
    size_t off = 0;
    auto alloc = [&](size_t n) { float* p = W + off; off += (n + 63) & ~(size_t)63; return p; };
    float* h      = alloc((size_t)N_NODES * HID);
    float* res    = alloc((size_t)N_NODES * HID);
    float* xl     = alloc((size_t)N_NODES * HID);
    float* xr     = alloc((size_t)N_NODES * HID);
    float* outb   = alloc((size_t)N_NODES * HID);
    float* sbuf   = alloc((size_t)EF * NHEAD);
    float* smax   = alloc((size_t)N_NODES * NHEAD);
    float* den    = alloc((size_t)N_NODES * NHEAD);
    float* deg    = alloc(N_NODES);
    float* lattr  = alloc((size_t)N_NODES * EDGE_DIM);
    float* stats  = alloc(4 * HID);
    float* pooled = alloc((size_t)NGRAPH * HID);
    float* cnt    = alloc(NGRAPH);
    bf16* hb  = (bf16*)alloc((size_t)N_NODES * HID / 2);
    bf16* Wlb = (bf16*)alloc((size_t)NLAYER * HID * HID / 2);
    bf16* Wrb = (bf16*)alloc((size_t)NLAYER * HID * HID / 2);

    const int NW = NLAYER * HID * HID;
    const int NH128 = (N_NODES * HID + 255) / 256;   // 8192 blocks
    const int EFB   = (EF + 255) / 256;
    const int EFHB  = (EF * NHEAD + 255) / 256;

    k_convert_weights<<<(NW + 255) / 256, 256, 0, stream>>>(Wl, Wr, Wlb, Wrb, NW);
    k_prep_init<<<(N_NODES * EDGE_DIM + 255) / 256, 256, 0, stream>>>(deg, lattr, pooled, cnt);
    k_deg_loop<<<(N_EDGES + 255) / 256, 256, 0, stream>>>(dst0, eattr, deg, lattr);
    k_loop_div<<<(N_NODES * EDGE_DIM + 255) / 256, 256, 0, stream>>>(lattr, deg);
    k_input_linear<<<NH128, 256, 0, stream>>>(x, Win, b_in, h);

    for (int l = 0; l < NLAYER; ++l) {
        k_h2bf16<<<NH128, 256, 0, stream>>>(h, hb, res, (l & 1) == 0);
        k_layer_init<<<NH128, 256, 0, stream>>>(outb, smax, den, stats);
        k_node_linear<<<N_NODES / 16, 256, 0, stream>>>(
            hb, Wlb + (size_t)l * HID * HID, Wrb + (size_t)l * HID * HID,
            bl + l * HID, br + l * HID, xl, xr);
        k_edge_score<<<EFB, 256, 0, stream>>>(
            src0, dst0, eattr, lattr, xl, xr,
            We + (size_t)l * EDGE_DIM * HID, att + (size_t)l * HID, sbuf, smax);
        k_edge_exp<<<EFHB, 256, 0, stream>>>(dst0, sbuf, smax, den);
        k_edge_aggr<<<EFHB, 256, 0, stream>>>(src0, dst0, sbuf, den, xl, outb);
        k_bn_stats<<<128, 128, 0, stream>>>(outb, stats);
        k_bn_finalize<<<1, 128, 0, stream>>>(stats);
        k_bn_apply<<<NH128, 256, 0, stream>>>(
            outb, stats, gamma + l * HID, beta + l * HID, res, h, l & 1);
    }

    k_pool_cnt<<<(N_NODES + 255) / 256, 256, 0, stream>>>(batch, cnt);
    k_pool_sum<<<NH128, 256, 0, stream>>>(batch, h, pooled);
    k_mlp<<<NGRAPH, 64, 0, stream>>>(pooled, cnt, W1, b1v, W2, b2v, (float*)d_out);
}
